// MultiHeadedSelfAttention_22488448761907
// MI455X (gfx1250) — matrix operand order repaired and checked by reading, not compiled, not run on hardware
//
#include <hip/hip_runtime.h>

typedef __bf16 bf16_t;
typedef __attribute__((ext_vector_type(16))) __bf16 v16bf;
typedef __attribute__((ext_vector_type(8)))  float  v8f;
typedef __attribute__((ext_vector_type(4)))  unsigned int v4u;
typedef __attribute__((ext_vector_type(8)))  unsigned int v8u;

constexpr int Bsz = 2;
constexpr int S   = 2048;
constexpr int E   = 1024;
constexpr int H   = 16;
constexpr int D   = 64;
constexpr int M   = Bsz * S;   // 4096 rows for the projection GEMMs

// ---------------------------------------------------------------------------
// WMMA fragment helpers (16x16x32 bf16, wave32; A and B share a contiguous k order, not the ISA table order)
// ---------------------------------------------------------------------------
__device__ __forceinline__ v8f zero8() {
    v8f z;
#pragma unroll
    for (int i = 0; i < 8; ++i) z[i] = 0.0f;
    return z;
}

// A-matrix 16x32 (MxK), 16-bit: lane L -> row m = L&15, koff = (L>>4)*16.
// halves 0..7  = A[m][koff + 0..7]
// halves 8..15 = A[m][koff + 8..15]
__device__ __forceinline__ v16bf load_frag_a(const bf16_t* p, int stride, int lane) {
    const int m    = lane & 15;
    const int koff = (lane >> 4) * 16;
    const bf16_t* row = p + m * stride;
    union { uint4 u[2]; v16bf v; } r;
    r.u[0] = *(const uint4*)(row + koff);
    r.u[1] = *(const uint4*)(row + koff + 8);
    return r.v;
}

// B-matrix 32x16 (KxN), 16-bit, stored as B^T row-major (row n, contiguous k):
// lane L -> col n = L&15, koff = (L>>4)*16; halves i = B[koff+i][n] = Bt[n][koff+i]
__device__ __forceinline__ v16bf load_frag_b(const bf16_t* p, int stride, int lane) {
    const int n    = lane & 15;
    const int koff = (lane >> 4) * 16;
    const bf16_t* row = p + n * stride + koff;
    union { uint4 u[2]; v16bf v; } r;
    r.u[0] = *(const uint4*)(row);
    r.u[1] = *(const uint4*)(row + 8);
    return r.v;
}

__device__ __forceinline__ v8f wmma_bf16(v16bf a, v16bf b, v8f c) {
    return __builtin_amdgcn_wmma_f32_16x16x32_bf16(false, a, false, b, (short)0, c,
                                                   false, false);
}

// ---------------------------------------------------------------------------
// TDM descriptor helpers (CDNA5 Tensor Data Mover, CDNA5 ISA)
// Group0: [1:0]=count, [63:32]=lds_addr, [120:64]=global_addr, [127:126]=type(2)
// Group1: [17:16]=data_size, [20]=pad_enable, [24:22]=pad_interval,
//         [31:25]=pad_amount, [79:48]=tensor_dim0, [111:80]=tensor_dim1,
//         [127:112]=tile_dim0, [143:128]=tile_dim1, [207:160]=tensor_dim0_stride
// ---------------------------------------------------------------------------
__device__ __forceinline__ v4u tdm_group0(unsigned lds_addr, const void* gptr) {
    unsigned long long ga = (unsigned long long)(uintptr_t)gptr;
    v4u g;
    g.x = 1u;                                   // count = 1 valid descriptor
    g.y = lds_addr;                             // LDS byte address
    g.z = (unsigned)ga;                         // global_addr[31:0]
    g.w = (unsigned)((ga >> 32) & 0x1FFFFFFu) | (2u << 30);  // ga[56:32] | type=2
    return g;
}

// 2D tile descriptor, 4-byte elements, LDS padding after every row.
__device__ __forceinline__ v8u tdm_group1(unsigned dim0_dw, unsigned rows,
                                          unsigned stride_dw, unsigned pad_ivl,
                                          unsigned pad_amt) {
    v8u g;
    g[0] = (2u << 16) | (1u << 20) | (pad_ivl << 22) | (pad_amt << 25);
    g[1] = (dim0_dw & 0xFFFFu) << 16;           // tensor_dim0 low16
    g[2] = ((dim0_dw >> 16) & 0xFFFFu) | ((rows & 0xFFFFu) << 16);
    g[3] = ((rows >> 16) & 0xFFFFu) | ((dim0_dw & 0xFFFFu) << 16);  // tile_dim0
    g[4] = rows & 0xFFFFu;                      // tile_dim1
    g[5] = stride_dw;                           // tensor_dim0_stride low32
    g[6] = (stride_dw >> 16) >> 16;             // (48-bit stride: high bits zero)
    g[7] = 0u;
    return g;
}

__device__ __forceinline__ void tdm_load(v4u g0, v8u g1) {
    asm volatile("tensor_load_to_lds %0, %1" :: "s"(g0), "s"(g1) : "memory");
}

// ---------------------------------------------------------------------------
// Prep kernels: f32 -> bf16 conversions / weight transposes
// ---------------------------------------------------------------------------
__global__ __launch_bounds__(256) void cvt_bf16_kernel(const float* __restrict__ in,
                                                       bf16_t* __restrict__ out, int n) {
    int i = blockIdx.x * 256 + threadIdx.x;
    if (i < n) out[i] = (bf16_t)in[i];
}

// W [H,E,D] -> Wt [(h*D+d)][E]  (B^T layout for GEMM)
__global__ __launch_bounds__(256) void transpose_hed_kernel(const float* __restrict__ w,
                                                            bf16_t* __restrict__ out) {
    int idx = blockIdx.x * 256 + threadIdx.x;   // over H*E*D = 1M
    if (idx >= H * E * D) return;
    int h = idx / (E * D);
    int r = idx % (E * D);
    int e = r / D;
    int d = r % D;
    out[(size_t)(h * D + d) * E + e] = (bf16_t)w[idx];
}

// Wo [E,E] -> WoT[n][k] = Wo[k][n]
__global__ __launch_bounds__(256) void transpose_ee_kernel(const float* __restrict__ w,
                                                           bf16_t* __restrict__ out) {
    int idx = blockIdx.x * 256 + threadIdx.x;   // over E*E
    if (idx >= E * E) return;
    int k = idx / E;
    int n = idx % E;
    out[(size_t)n * E + k] = (bf16_t)w[idx];
}

// ---------------------------------------------------------------------------
// Fused QKV projection GEMM.
// C[m][n] = sum_e x[m][e] * W[e][n] + bias[n],  m = b*S+s, n = h*D+d
// blockIdx.z selects Q (0), K (1), V (2).  Q,K stored [B,H,S,D] bf16;
// V stored transposed [B,H,D,S] bf16 for the attention PV matmul.
// ---------------------------------------------------------------------------
__global__ __launch_bounds__(256) void qkv_gemm_kernel(
    const bf16_t* __restrict__ xb,
    const bf16_t* __restrict__ WtQ, const bf16_t* __restrict__ WtK,
    const bf16_t* __restrict__ WtV,
    const float* __restrict__ bq, const float* __restrict__ bk,
    const float* __restrict__ bv,
    bf16_t* __restrict__ Qo, bf16_t* __restrict__ Ko, bf16_t* __restrict__ Vto) {
    __shared__ __attribute__((aligned(16))) bf16_t As[128][40];
    __shared__ __attribute__((aligned(16))) bf16_t Bs[64][40];

    const int tid  = threadIdx.x;
    const int lane = tid & 31;
    const int wave = tid >> 5;
    const int z    = blockIdx.z;

    const bf16_t* Wt   = (z == 0) ? WtQ : (z == 1) ? WtK : WtV;
    const float*  bias = (z == 0) ? bq : (z == 1) ? bk : bv;

    const int m0 = blockIdx.y * 128;
    const int n0 = blockIdx.x * 64;
    const int wm = (wave >> 1) * 32;
    const int wn = (wave & 1) * 32;

    v8f acc[2][2];
#pragma unroll
    for (int i = 0; i < 2; ++i)
#pragma unroll
        for (int j = 0; j < 2; ++j) acc[i][j] = zero8();

    const int arow = tid >> 1, acol = (tid & 1) * 16;   // 128 rows x 32 cols
    const int brow = tid >> 2, bcol = (tid & 3) * 8;    // 64 rows x 32 cols

    for (int k0 = 0; k0 < E; k0 += 32) {
        const uint4* ag = (const uint4*)(xb + (size_t)(m0 + arow) * E + k0 + acol);
        *(uint4*)&As[arow][acol]     = ag[0];
        *(uint4*)&As[arow][acol + 8] = ag[1];
        const uint4* bg = (const uint4*)(Wt + (size_t)(n0 + brow) * E + k0 + bcol);
        *(uint4*)&Bs[brow][bcol] = bg[0];
        __syncthreads();

        v16bf a0 = load_frag_a(&As[wm][0],      40, lane);
        v16bf a1 = load_frag_a(&As[wm + 16][0], 40, lane);
        v16bf b0 = load_frag_b(&Bs[wn][0],      40, lane);
        v16bf b1 = load_frag_b(&Bs[wn + 16][0], 40, lane);

        acc[0][0] = wmma_bf16(a0, b0, acc[0][0]);
        acc[0][1] = wmma_bf16(a0, b1, acc[0][1]);
        acc[1][0] = wmma_bf16(a1, b0, acc[1][0]);
        acc[1][1] = wmma_bf16(a1, b1, acc[1][1]);
        __syncthreads();
    }

    // Epilogue. C layout: lane -> col n = lane&15; VGPR r -> row r (lanes<16) / 8+r.
    const int hi = (lane >> 4);
#pragma unroll
    for (int mt = 0; mt < 2; ++mt) {
#pragma unroll
        for (int nt = 0; nt < 2; ++nt) {
            const int n  = n0 + wn + nt * 16 + (lane & 15);
            const int h  = n >> 6;           // n / D
            const int d  = n & 63;           // n % D
            const float bb = bias[n];
            const int mbase = m0 + wm + mt * 16 + (hi ? 8 : 0);
#pragma unroll
            for (int r = 0; r < 8; ++r) {
                const int gm = mbase + r;
                const int bi = gm >> 11;     // / S
                const int s  = gm & 2047;    // % S
                const float val = acc[mt][nt][r] + bb;
                if (z == 2)
                    Vto[(((size_t)bi * H + h) * D + d) * S + s] = (bf16_t)val;
                else if (z == 0)
                    Qo[(((size_t)bi * H + h) * S + s) * D + d] = (bf16_t)val;
                else
                    Ko[(((size_t)bi * H + h) * S + s) * D + d] = (bf16_t)val;
            }
        }
    }
}

// ---------------------------------------------------------------------------
// Flash attention with TDM double-buffered K/V staging.
// S^T = K * Q^T per tile (softmax reductions ~lane-local), ctx^T = V^T * P^T.
// Each wave owns 16 query rows; block = 8 waves = 128 rows.
// Wave 0 drives the Tensor Data Mover; tiles land in LDS with hardware row
// padding (K: 128B+16B -> pitch 72 elem, V: 64B+16B -> pitch 40 elem).
// ---------------------------------------------------------------------------
__global__ __launch_bounds__(256) void attn_kernel(
    const bf16_t* __restrict__ Q, const bf16_t* __restrict__ K,
    const bf16_t* __restrict__ Vt, const int* __restrict__ mask,
    bf16_t* __restrict__ ctx) {
    __shared__ __attribute__((aligned(16))) bf16_t Ks[2][32][72];   // [t][d]
    __shared__ __attribute__((aligned(16))) bf16_t Vs[2][64][40];   // [d][t]

    const int tid  = threadIdx.x;
    const int lane = tid & 31;
    const int wave = tid >> 5;
    const int bh   = blockIdx.y;          // b*H + h
    const int b    = bh >> 4;             // / H
    const int h    = bh & 15;             // % H
    const bool hi  = (lane >= 16);

    const int s_base = blockIdx.x * 128 + wave * 16;
    const int s_lane = s_base + (lane & 15);

    // Persistent Q^T B-fragments: lane holds row s_lane, contiguous d.
    v16bf Qf[2];
    {
        const bf16_t* qrow = Q + ((size_t)bh * S + s_lane) * D + ((lane >> 4) * 16);
        union { uint4 u[2]; v16bf v; } r0, r1;
        r0.u[0] = *(const uint4*)(qrow);
        r0.u[1] = *(const uint4*)(qrow + 8);
        r1.u[0] = *(const uint4*)(qrow + 32);
        r1.u[1] = *(const uint4*)(qrow + 40);
        Qf[0] = r0.v;
        Qf[1] = r1.v;
    }

    const int* mrow = mask + (size_t)b * S * S + (size_t)s_lane * S;

    float m_run = -3.0e38f;
    float l_run = 0.0f;
    v8f O[4];
#pragma unroll
    for (int i = 0; i < 4; ++i) O[i] = zero8();

    // TDM group-1 descriptors (uniform):
    // K tile: 32 rows x 32 dwords (128B), row stride 32 dw, pad 4dw after 32dw
    //   pad_interval: 32dw = 16 x 8B units -> code 4 ; pad_amount: 4dw -> code 3
    const v8u g1k = tdm_group1(32u, 32u, 32u, 4u, 3u);
    // V tile: 64 rows x 16 dwords (64B), row stride 1024 dw, pad 4dw after 16dw
    //   pad_interval: 16dw = 8 x 8B units -> code 3 ; pad_amount: 4dw -> code 3
    const v8u g1v = tdm_group1(16u, 64u, 1024u, 3u, 3u);

    const bf16_t* Kbh  = K  + (size_t)bh * S * D;
    const bf16_t* Vtbh = Vt + (size_t)bh * D * S;

    constexpr int nIter = S / 32;   // 64 K/V tiles

    // prologue: DMA tile 0 into buffer 0
    if (wave == 0) {
        tdm_load(tdm_group0((unsigned)(uintptr_t)&Ks[0][0][0], Kbh), g1k);
        tdm_load(tdm_group0((unsigned)(uintptr_t)&Vs[0][0][0], Vtbh), g1v);
    }

    for (int it = 0; it < nIter; ++it) {
        const int buf = it & 1;
        const int t0  = it * 32;

        if (wave == 0) {
            if (it + 1 < nIter) {
                // issue next tile into the other buffer, then wait for current
                // (TDM ops complete in order: <=2 outstanding => tile `it` done)
                const int nb = buf ^ 1;
                tdm_load(tdm_group0((unsigned)(uintptr_t)&Ks[nb][0][0],
                                    Kbh + (size_t)(t0 + 32) * D), g1k);
                tdm_load(tdm_group0((unsigned)(uintptr_t)&Vs[nb][0][0],
                                    Vtbh + (t0 + 32)), g1v);
                __builtin_amdgcn_s_wait_tensorcnt(2);
            } else {
                __builtin_amdgcn_s_wait_tensorcnt(0);
            }
        }
        __syncthreads();

        // S^T tiles: [t][s], 2 tiles of 16 t-rows covering Bc=32
        v8f st[2];
#pragma unroll
        for (int tt = 0; tt < 2; ++tt) {
            v16bf kA0 = load_frag_a(&Ks[buf][tt * 16][0],  72, lane);
            v16bf kA1 = load_frag_a(&Ks[buf][tt * 16][32], 72, lane);
            st[tt] = zero8();
            st[tt] = wmma_bf16(kA0, Qf[0], st[tt]);
            st[tt] = wmma_bf16(kA1, Qf[1], st[tt]);
        }

        // scale + mask ( -1e9 like reference ), local row max over this tile
        float sc[2][8];
        float mloc = -3.0e38f;
#pragma unroll
        for (int tt = 0; tt < 2; ++tt) {
            const int* mp = mrow + t0 + tt * 16 + (hi ? 8 : 0);
            int mk[8];
            *(int4*)&mk[0] = *(const int4*)(mp);
            *(int4*)&mk[4] = *(const int4*)(mp + 4);
#pragma unroll
            for (int r = 0; r < 8; ++r) {
                float v = (mk[r] != 0) ? st[tt][r] * 0.125f : -1.0e9f;
                sc[tt][r] = v;
                mloc = fmaxf(mloc, v);
            }
        }
        // combine the two lane-halves holding the same s column
        mloc = fmaxf(mloc, __shfl_xor(mloc, 16, 32));

        const float m_new = fmaxf(m_run, mloc);
        const float corr  = __expf(m_run - m_new);

        float p[2][8];
        float lsum = 0.0f;
#pragma unroll
        for (int tt = 0; tt < 2; ++tt)
#pragma unroll
            for (int r = 0; r < 8; ++r) {
                float e = __expf(sc[tt][r] - m_new);
                p[tt][r] = e;
                lsum += e;
            }
        lsum += __shfl_xor(lsum, 16, 32);
        l_run = l_run * corr + lsum;
        m_run = m_new;

#pragma unroll
        for (int dt = 0; dt < 4; ++dt)
#pragma unroll
            for (int e = 0; e < 8; ++e) O[dt][e] *= corr;

        // Assemble P^T B-fragment (k = t over 0..31, n = s): one cross-half shuffle
        v16bf pf;
#pragma unroll
        for (int r = 0; r < 8; ++r) {
            float o0 = __shfl_xor(p[0][r], 16, 32);   // partner's tile0
            float o1 = __shfl_xor(p[1][r], 16, 32);   // partner's tile1
            float lo = hi ? o1 : p[0][r];             // half r     : t = koff + r
            float hv = hi ? p[1][r] : o0;             // half 8 + r : t = koff + 8 + r
            pf[r]     = (bf16_t)lo;
            pf[8 + r] = (bf16_t)hv;
        }

        // ctx^T += V^T * P^T  (4 d-tiles of 16)
#pragma unroll
        for (int dt = 0; dt < 4; ++dt) {
            v16bf vA = load_frag_a(&Vs[buf][dt * 16][0], 40, lane);
            O[dt] = wmma_bf16(vA, pf, O[dt]);
        }
        __syncthreads();   // done reading buf before TDM overwrites it (iter it+1)
    }

    // Finalize & store ctx[b][s][h*D + d] as bf16 (packed dword stores: d pairs)
    const float inv = 1.0f / l_run;
    bf16_t* crow = ctx + ((size_t)b * S + s_lane) * E + h * D;
#pragma unroll
    for (int dt = 0; dt < 4; ++dt) {
        const int dbase = dt * 16 + (hi ? 8 : 0);
#pragma unroll
        for (int j = 0; j < 4; ++j) {
            union { bf16_t h2[2]; unsigned u; } pk;
            pk.h2[0] = (bf16_t)(O[dt][2 * j] * inv);
            pk.h2[1] = (bf16_t)(O[dt][2 * j + 1] * inv);
            *(unsigned*)(crow + dbase + 2 * j) = pk.u;
        }
    }
}

// ---------------------------------------------------------------------------
// Output projection: out[m][n] = sum_k ctx[m][k] * Wo[k][n] + bo[n], f32 out.
// ---------------------------------------------------------------------------
__global__ __launch_bounds__(256) void out_gemm_kernel(
    const bf16_t* __restrict__ cb, const bf16_t* __restrict__ WoT,
    const float* __restrict__ bo, float* __restrict__ out) {
    __shared__ __attribute__((aligned(16))) bf16_t As[128][40];
    __shared__ __attribute__((aligned(16))) bf16_t Bs[64][40];

    const int tid  = threadIdx.x;
    const int lane = tid & 31;
    const int wave = tid >> 5;

    const int m0 = blockIdx.y * 128;
    const int n0 = blockIdx.x * 64;
    const int wm = (wave >> 1) * 32;
    const int wn = (wave & 1) * 32;

    v8f acc[2][2];
#pragma unroll
    for (int i = 0; i < 2; ++i)
#pragma unroll
        for (int j = 0; j < 2; ++j) acc[i][j] = zero8();

    const int arow = tid >> 1, acol = (tid & 1) * 16;
    const int brow = tid >> 2, bcol = (tid & 3) * 8;

    for (int k0 = 0; k0 < E; k0 += 32) {
        const uint4* ag = (const uint4*)(cb + (size_t)(m0 + arow) * E + k0 + acol);
        *(uint4*)&As[arow][acol]     = ag[0];
        *(uint4*)&As[arow][acol + 8] = ag[1];
        const uint4* bg = (const uint4*)(WoT + (size_t)(n0 + brow) * E + k0 + bcol);
        *(uint4*)&Bs[brow][bcol] = bg[0];
        __syncthreads();

        v16bf a0 = load_frag_a(&As[wm][0],      40, lane);
        v16bf a1 = load_frag_a(&As[wm + 16][0], 40, lane);
        v16bf b0 = load_frag_b(&Bs[wn][0],      40, lane);
        v16bf b1 = load_frag_b(&Bs[wn + 16][0], 40, lane);

        acc[0][0] = wmma_bf16(a0, b0, acc[0][0]);
        acc[0][1] = wmma_bf16(a0, b1, acc[0][1]);
        acc[1][0] = wmma_bf16(a1, b0, acc[1][0]);
        acc[1][1] = wmma_bf16(a1, b1, acc[1][1]);
        __syncthreads();
    }

    const int hi = (lane >> 4);
#pragma unroll
    for (int mt = 0; mt < 2; ++mt) {
#pragma unroll
        for (int nt = 0; nt < 2; ++nt) {
            const int n  = n0 + wn + nt * 16 + (lane & 15);
            const float bb = bo[n];
            const int mbase = m0 + wm + mt * 16 + (hi ? 8 : 0);
#pragma unroll
            for (int r = 0; r < 8; ++r) {
                const int gm = mbase + r;
                out[(size_t)gm * E + n] = acc[mt][nt][r] + bb;
            }
        }
    }
}

// ---------------------------------------------------------------------------
extern "C" void kernel_launch(void* const* d_in, const int* in_sizes, int n_in,
                              void* d_out, int out_size, void* d_ws, size_t ws_size,
                              hipStream_t stream) {
    (void)in_sizes; (void)n_in; (void)out_size;
    const float* x    = (const float*)d_in[0];
    const int*   mask = (const int*)d_in[1];
    const float* Wq   = (const float*)d_in[2];
    const float* bq   = (const float*)d_in[3];
    const float* Wk   = (const float*)d_in[4];
    const float* bk   = (const float*)d_in[5];
    const float* Wv   = (const float*)d_in[6];
    const float* bv   = (const float*)d_in[7];
    const float* Wo   = (const float*)d_in[8];
    const float* bo   = (const float*)d_in[9];
    float* out = (float*)d_out;

    // workspace carve-up (bf16 buffers)
    char* w = (char*)d_ws;
    const size_t szME = (size_t)M * E * 2;      // 8 MB
    const size_t szEE = (size_t)E * E * 2;      // 2 MB
    bf16_t* xb  = (bf16_t*)w;              w += szME;
    bf16_t* WtQ = (bf16_t*)w;              w += szEE;
    bf16_t* WtK = (bf16_t*)w;              w += szEE;
    bf16_t* WtV = (bf16_t*)w;              w += szEE;
    bf16_t* WoT = (bf16_t*)w;              w += szEE;
    bf16_t* Qb  = (bf16_t*)w;              w += szME;
    bf16_t* Kb  = (bf16_t*)w;              w += szME;
    bf16_t* Vtb = (bf16_t*)w;              w += szME;
    bf16_t* cb  = (bf16_t*)w;              w += szME;
    if ((size_t)(w - (char*)d_ws) > ws_size) return;   // insufficient scratch

    // 1) conversions / transposes
    cvt_bf16_kernel<<<(M * E + 255) / 256, 256, 0, stream>>>(x, xb, M * E);
    transpose_hed_kernel<<<(H * E * D + 255) / 256, 256, 0, stream>>>(Wq, WtQ);
    transpose_hed_kernel<<<(H * E * D + 255) / 256, 256, 0, stream>>>(Wk, WtK);
    transpose_hed_kernel<<<(H * E * D + 255) / 256, 256, 0, stream>>>(Wv, WtV);
    transpose_ee_kernel<<<(E * E + 255) / 256, 256, 0, stream>>>(Wo, WoT);

    // 2) fused QKV projection (z = Q/K/V)
    qkv_gemm_kernel<<<dim3(E / 64, M / 128, 3), 256, 0, stream>>>(
        xb, WtQ, WtK, WtV, bq, bk, bv, Qb, Kb, Vtb);

    // 3) flash attention -> ctx (concat layout, bf16)
    attn_kernel<<<dim3(S / 128, Bsz * H), 256, 0, stream>>>(Qb, Kb, Vtb, mask, cb);

    // 4) output projection -> f32 out
    out_gemm_kernel<<<dim3(E / 64, M / 128), 256, 0, stream>>>(cb, WoT, bo, out);
}
